// EncoderBlock_40020505264744
// MI455X (gfx1250) — matrix operand order repaired and checked by reading, not compiled, not run on hardware
//
#include <hip/hip_runtime.h>

// ---------------- constants ----------------
#define C_DIM   512
#define WSZ     8
#define NHEAD   8
#define HDIM    64
#define M_ROWS  32          // 4 windows * 8 tokens per workgroup
#define THREADS 512         // 16 waves -> 4 waves/SIMD
#define STA     520         // bf16 row stride (elements) for A tile
#define STQ     1544        // bf16 row stride for QKV tile
#define STY     520         // f32 row stride for Y tile
#define KTILES  16          // 512 / 32

// d_ws layout: bf16 weights then f32 params, single base pointer (keeps
// every parameter access in addrspace(1) -> global_load, never flat_load).
#define OFF_Q1  0
#define OFF_P1  786432      // 1536*512
#define OFF_Q2  1048576     // + 512*512
#define OFF_P2  1835008     // + 1536*512
#define W_TOTAL 2097152     // ushort count of bf16 weights
// f32 param region (float offsets relative to wp + W_TOTAL):
#define FB_Q1B  0           // 1536
#define FB_P1B  1536        // 512
#define FB_Q2B  2048        // 1536
#define FB_P2B  3584        // 512
#define FB_L1W  4096        // 4096
#define FB_L1B  8192        // 4096
#define FB_L3W  12288       // 4096
#define FB_L3B  16384       // 4096
#define F_TOTAL 20480

typedef __attribute__((ext_vector_type(16))) __bf16 v16bf;
typedef __attribute__((ext_vector_type(2)))  __bf16 v2bf;
typedef __attribute__((ext_vector_type(8)))  float  v8f;

static __device__ __forceinline__ unsigned short f2bf(float f) {
    union { float f; unsigned u; } x; x.f = f;
    unsigned u = x.u;
    u += 0x7fffu + ((u >> 16) & 1u);          // round-to-nearest-even
    return (unsigned short)(u >> 16);
}
static __device__ __forceinline__ unsigned f2bf_pk(float a, float b) {
#if __has_builtin(__builtin_amdgcn_cvt_pk_bf16_f32)
    union { v2bf v; unsigned u; } r;
    r.v = __builtin_amdgcn_cvt_pk_bf16_f32(a, b);
    return r.u;
#else
    return (unsigned)f2bf(a) | ((unsigned)f2bf(b) << 16);
#endif
}
static __device__ __forceinline__ float bf2f(unsigned short h) {
    union { unsigned u; float f; } x; x.u = ((unsigned)h) << 16;
    return x.f;
}

union FragU { v16bf v; uint4 q[2]; };

// A fragment 16x32 in this code's contiguous k order (matches B): lane<16 holds M=lane, K={0..15};
// lane>=16 holds M=lane-16, K={16..31}.  Two ds_load_b128 per fragment.
static __device__ __forceinline__ v16bf load_a_frag(const unsigned short* sA,
                                                    int m0, int k0, int lane) {
    int row = m0 + (lane & 15);
    int kk  = k0 + ((lane >> 4) << 4);
    FragU u;
    u.q[0] = *reinterpret_cast<const uint4*>(sA + row * STA + kk);
    u.q[1] = *reinterpret_cast<const uint4*>(sA + row * STA + kk + 8);
    return u.v;
}

// B-matrix 32x16: lane holds column n0+(lane&15), 16 contiguous K values
// -> 32 contiguous bytes of the row-major [Nout][K] bf16 matrix (L2-hot).
static __device__ __forceinline__ v16bf load_b_frag(const unsigned short* w,
                                                    int n0, int k0, int lane) {
    int n  = n0 + (lane & 15);
    int kk = k0 + ((lane >> 4) << 4);
    FragU u;
    const uint4* p = reinterpret_cast<const uint4*>(w + n * C_DIM + kk);
    u.q[0] = p[0];
    u.q[1] = p[1];
    return u.v;
}

// One 16x16 output tile, K=512 contraction via 16 x v_wmma_f32_16x16x32_bf16
static __device__ __forceinline__ v8f gemm_tile(const unsigned short* sA,
                                                const unsigned short* w,
                                                int m0, int n0, int lane,
                                                float bias) {
    v8f acc;
#pragma unroll
    for (int r = 0; r < 8; ++r) acc[r] = bias;
#pragma unroll 4
    for (int kk = 0; kk < KTILES; ++kk) {
        v16bf a = load_a_frag(sA, m0, kk * 32, lane);
        v16bf b = load_b_frag(w,  n0, kk * 32, lane);
        acc = __builtin_amdgcn_wmma_f32_16x16x32_bf16(
            false, a, false, b, (short)0, acc, false, false);
    }
    return acc;
}

// ---------------- parameter pre-pack (weights->bf16, params->f32 region) ----
__global__ void pack_params(const float* __restrict__ q1w, const float* __restrict__ p1w,
                            const float* __restrict__ q2w, const float* __restrict__ p2w,
                            const float* __restrict__ q1b, const float* __restrict__ p1b,
                            const float* __restrict__ q2b, const float* __restrict__ p2b,
                            const float* __restrict__ l1w, const float* __restrict__ l1b,
                            const float* __restrict__ l3w, const float* __restrict__ l3b,
                            unsigned short* __restrict__ wp) {
    int i = blockIdx.x * 256 + threadIdx.x;
    if (i < W_TOTAL) {
        float v;
        if      (i < OFF_P1) v = q1w[i];
        else if (i < OFF_Q2) v = p1w[i - OFF_P1];
        else if (i < OFF_P2) v = q2w[i - OFF_Q2];
        else                 v = p2w[i - OFF_P2];
        wp[i] = f2bf(v);
    } else {
        int j = i - W_TOTAL;
        if (j >= F_TOTAL) return;
        float v;
        if      (j < FB_P1B) v = q1b[j - FB_Q1B];
        else if (j < FB_Q2B) v = p1b[j - FB_P1B];
        else if (j < FB_P2B) v = q2b[j - FB_Q2B];
        else if (j < FB_L1W) v = p2b[j - FB_P2B];
        else if (j < FB_L1B) v = l1w[j - FB_L1W];
        else if (j < FB_L3W) v = l1b[j - FB_L1B];
        else if (j < FB_L3B) v = l3w[j - FB_L3W];
        else                 v = l3b[j - FB_L3B];
        reinterpret_cast<float*>(wp + W_TOTAL)[j] = v;
    }
}

// ---------------- fused two-layer windowed attention ----------------
__global__ __launch_bounds__(THREADS, 1)
void encoder_fused(const float* __restrict__ x,
                   const unsigned short* __restrict__ wp,
                   float* __restrict__ out) {
    extern __shared__ char smem[];
    unsigned short* sA   = (unsigned short*)smem;                  // [32][520] bf16
    unsigned short* sQKV = (unsigned short*)(smem + 33280);        // [32][1544] bf16
    float*          sY   = (float*)(smem + 33280 + 98816);         // [32][520] f32
    __shared__ float redS[16];
    __shared__ float redQ[16];

    const int tid  = threadIdx.x;
    const int lane = tid & 31;
    const int wave = tid >> 5;
    const size_t rowBase = (size_t)blockIdx.x * M_ROWS;
    const float* fpar = reinterpret_cast<const float*>(wp + W_TOTAL);

    // ---- stage x (4 windows, [32,512] f32) -> sY via async-to-LDS DMA ----
    {
        const float* xg = x + rowBase * C_DIM;
#pragma unroll 4
        for (int i = tid; i < (M_ROWS * C_DIM) / 4; i += THREADS) {
            int flat = i << 2;
            int r = flat >> 9, c = flat & (C_DIM - 1);
            unsigned ldsoff = (unsigned)(size_t)(&sY[r * STY + c]);
            unsigned long long ga = (unsigned long long)(size_t)(xg + flat);
            asm volatile("global_load_async_to_lds_b128 %0, %1, off"
                         :: "v"(ldsoff), "v"(ga) : "memory");
        }
        asm volatile("s_wait_asynccnt 0" ::: "memory");
    }
    __syncthreads();

    for (int s = 0; s < 2; ++s) {
        const unsigned short* qkvW  = wp + (s ? OFF_Q2 : OFF_Q1);
        const unsigned short* projW = wp + (s ? OFF_P2 : OFF_P1);
        const float* qkvB = fpar + (s ? FB_Q2B : FB_Q1B);
        const float* projB = fpar + (s ? FB_P2B : FB_P1B);
        const float* lnWp = fpar + (s ? FB_L3W : FB_L1W);
        const float* lnBp = fpar + (s ? FB_L3B : FB_L1B);

        // ---- LayerNorm over each window [8,512] : sY -> sA (bf16) ----
        {
            const int w = tid >> 7;      // window in group (4 waves per window)
            const int t = tid & 127;
            float sm = 0.f, sq = 0.f;
#pragma unroll 8
            for (int j = 0; j < 32; ++j) {
                int e = t + (j << 7);
                int r = e >> 9, c = e & 511;
                float v = sY[(w * WSZ + r) * STY + c];
                sm += v; sq += v * v;
            }
#pragma unroll
            for (int off = 16; off > 0; off >>= 1) {
                sm += __shfl_xor(sm, off, 32);
                sq += __shfl_xor(sq, off, 32);
            }
            if (lane == 0) { redS[wave] = sm; redQ[wave] = sq; }
            __syncthreads();
            float tot = redS[w*4] + redS[w*4+1] + redS[w*4+2] + redS[w*4+3];
            float tsq = redQ[w*4] + redQ[w*4+1] + redQ[w*4+2] + redQ[w*4+3];
            const float inv_n = 1.0f / 4096.0f;
            float mu   = tot * inv_n;
            float var  = tsq * inv_n - mu * mu;
            float rinv = rsqrtf(var + 1e-5f);
#pragma unroll 8
            for (int j = 0; j < 32; ++j) {
                int e = t + (j << 7);
                int r = e >> 9, c = e & 511;
                float v  = sY[(w * WSZ + r) * STY + c];
                float nv = (v - mu) * rinv * lnWp[r * C_DIM + c] + lnBp[r * C_DIM + c];
                sA[(w * WSZ + r) * STA + c] = f2bf(nv);
            }
        }
        __syncthreads();

        // ---- QKV GEMM: [32,512] x [512,1536] -> sQKV (q pre-scaled) ----
        for (int tle = wave; tle < 192; tle += 16) {     // 2 Mtiles x 96 Ntiles
            int m0 = (tle & 1) << 4;
            int n0 = (tle >> 1) << 4;
            float bias = qkvB[n0 + (lane & 15)];
            v8f acc = gemm_tile(sA, qkvW, m0, n0, lane, bias);
            float scale = (n0 < C_DIM) ? 0.125f : 1.0f;  // HD^-0.5 on Q block
            int col = n0 + (lane & 15);
            int rb  = m0 + ((lane >> 4) << 3);
#pragma unroll
            for (int r = 0; r < 8; ++r)
                sQKV[(rb + r) * STQ + col] = f2bf(acc[r] * scale);
        }
        __syncthreads();

        // ---- attention: lane-pair per (window, head, q-row); HD split in 2 ----
        {
            const int w    = tid >> 7;
            const int h    = (tid >> 4) & 7;
            const int qr   = (tid >> 1) & 7;
            const int half = tid & 1;
            const int hoff = h * HDIM + half * 32;
            const unsigned short* qrow = sQKV + (w * WSZ + qr) * STQ + hoff;
            float qv[32];
#pragma unroll
            for (int i = 0; i < 32; ++i) qv[i] = bf2f(qrow[i]);
            float sc[WSZ];
#pragma unroll
            for (int kr = 0; kr < WSZ; ++kr) {
                const unsigned short* krow =
                    sQKV + (w * WSZ + kr) * STQ + C_DIM + hoff;
                float d = 0.f;
#pragma unroll
                for (int i = 0; i < 32; ++i) d += qv[i] * bf2f(krow[i]);
                d += __shfl_xor(d, 1, 32);               // combine the two halves
                sc[kr] = d;
            }
            float m = sc[0];
#pragma unroll
            for (int kr = 1; kr < WSZ; ++kr) m = fmaxf(m, sc[kr]);
            float ssum = 0.f;
#pragma unroll
            for (int kr = 0; kr < WSZ; ++kr) { sc[kr] = __expf(sc[kr] - m); ssum += sc[kr]; }
            float inv = 1.0f / ssum;
            float o[32];
#pragma unroll
            for (int i = 0; i < 32; ++i) o[i] = 0.f;
#pragma unroll
            for (int kr = 0; kr < WSZ; ++kr) {
                const unsigned short* vrow =
                    sQKV + (w * WSZ + kr) * STQ + 2 * C_DIM + hoff;
                float p = sc[kr];
#pragma unroll
                for (int i = 0; i < 32; ++i) o[i] += p * bf2f(vrow[i]);
            }
            unsigned* orow = reinterpret_cast<unsigned*>(
                sA + (w * WSZ + qr) * STA + hoff);
#pragma unroll
            for (int i = 0; i < 32; i += 2)
                orow[i >> 1] = f2bf_pk(o[i] * inv, o[i + 1] * inv);
        }
        __syncthreads();

        // ---- proj GEMM: [32,512] x [512,512]; s==0 -> sY, s==1 -> global ----
        for (int tle = wave; tle < 64; tle += 16) {      // 2 Mtiles x 32 Ntiles
            int m0 = (tle & 1) << 4;
            int n0 = (tle >> 1) << 4;
            float bias = projB[n0 + (lane & 15)];
            v8f acc = gemm_tile(sA, projW, m0, n0, lane, bias);
            int col = n0 + (lane & 15);
            int rb  = m0 + ((lane >> 4) << 3);
            if (s == 0) {
#pragma unroll
                for (int r = 0; r < 8; ++r)
                    sY[(rb + r) * STY + col] = acc[r];
            } else {
#pragma unroll
                for (int r = 0; r < 8; ++r)
                    out[(rowBase + rb + r) * C_DIM + col] = acc[r];
            }
        }
        __syncthreads();
    }
}

extern "C" void kernel_launch(void* const* d_in, const int* in_sizes, int n_in,
                              void* d_out, int out_size, void* d_ws, size_t ws_size,
                              hipStream_t stream) {
    (void)in_sizes; (void)n_in; (void)out_size; (void)ws_size;
    const float* x      = (const float*)d_in[0];
    const float* ln1w   = (const float*)d_in[1];
    const float* ln1b   = (const float*)d_in[2];
    const float* ln3w   = (const float*)d_in[3];
    const float* ln3b   = (const float*)d_in[4];
    const float* qkv1w  = (const float*)d_in[5];
    const float* qkv1b  = (const float*)d_in[6];
    const float* proj1w = (const float*)d_in[7];
    const float* proj1b = (const float*)d_in[8];
    const float* qkv2w  = (const float*)d_in[9];
    const float* qkv2b  = (const float*)d_in[10];
    const float* proj2w = (const float*)d_in[11];
    const float* proj2b = (const float*)d_in[12];
    float* out = (float*)d_out;
    unsigned short* wpk = (unsigned short*)d_ws;   // ~4.3 MB packed params

    const int n_pack = (W_TOTAL + F_TOTAL + 255) / 256;
    pack_params<<<n_pack, 256, 0, stream>>>(qkv1w, proj1w, qkv2w, proj2w,
                                            qkv1b, proj1b, qkv2b, proj2b,
                                            ln1w, ln1b, ln3w, ln3b, wpk);

    const int smem = 33280 + 98816 + 66560;       // 198,656 B < 320 KB/WGP
    const int n_groups = (16 * 8192 / WSZ) / 4;   // 16384 windows / 4 = 4096 WGs
    encoder_fused<<<n_groups, THREADS, smem, stream>>>(x, wpk, out);
}
